// OwnBinaryConv_72894184948222
// MI455X (gfx1250) — 8-bit matrix operand order repaired and checked by reading, compiled, not run on hardware
//
#include <hip/hip_runtime.h>
#include <hip/hip_bf16.h>
#include <stdint.h>

typedef __attribute__((ext_vector_type(8)))  float        v8f;
typedef __attribute__((ext_vector_type(16))) int          v16i;
typedef __attribute__((ext_vector_type(4)))  unsigned int v4u;
typedef __attribute__((ext_vector_type(8)))  int          v8i;
typedef __attribute__((ext_vector_type(4)))  int          v4i;

#define NB      32
#define ICH     256
#define OCH     256
#define HH      56
#define HWP     58
#define HWPIX   3136
#define XPAD_BYTES (32u*58u*58u*256u)
#define WF8_BYTES  (9u*256u*256u)
#define ROWS    144

#if __has_builtin(__builtin_amdgcn_tensor_load_to_lds)
#define USE_TDM 1
#else
#define USE_TDM 0
#endif

__device__ __forceinline__ unsigned char sgn_fp8(float v) {
    return v > 0.f ? 0x38 : (v < 0.f ? 0xB8 : 0x00);
}

__global__ void bz_zero_kernel(uint4* __restrict__ p, unsigned int n16) {
    unsigned int i = blockIdx.x * 256u + threadIdx.x;
    if (i < n16) { uint4 z = {0u, 0u, 0u, 0u}; p[i] = z; }
}

__global__ void bz_pack_x_kernel(const float* __restrict__ x,
                                 const float* __restrict__ sh1,
                                 const float* __restrict__ sh2,
                                 unsigned char* __restrict__ xp1,
                                 unsigned char* __restrict__ xp2) {
    __shared__ uint32_t sl[2][HH * 64];
    const int n  = blockIdx.x / HH;
    const int y  = blockIdx.x % HH;
    const int ic = threadIdx.x;
    const float s1 = sh1[ic], s2 = sh2[ic];
    const float4* row =
        (const float4*)(x + (((size_t)(n * ICH + ic) * HH) + y) * HH);
    unsigned char* b1 = (unsigned char*)&sl[0][0];
    unsigned char* b2 = (unsigned char*)&sl[1][0];
#pragma unroll
    for (int q = 0; q < 14; ++q) {
        float4 f = row[q];
        int xx = q * 4;
        b1[(xx + 0) * 256 + ic] = sgn_fp8(f.x + s1);
        b1[(xx + 1) * 256 + ic] = sgn_fp8(f.y + s1);
        b1[(xx + 2) * 256 + ic] = sgn_fp8(f.z + s1);
        b1[(xx + 3) * 256 + ic] = sgn_fp8(f.w + s1);
        b2[(xx + 0) * 256 + ic] = sgn_fp8(f.x + s2);
        b2[(xx + 1) * 256 + ic] = sgn_fp8(f.y + s2);
        b2[(xx + 2) * 256 + ic] = sgn_fp8(f.z + s2);
        b2[(xx + 3) * 256 + ic] = sgn_fp8(f.w + s2);
    }
    __syncthreads();
    for (int j = threadIdx.x; j < 2 * HH * 64; j += 256) {
        const int v   = j / (HH * 64);
        const int rem = j % (HH * 64);
        const int xi  = rem >> 6;
        const int dw  = rem & 63;
        uint32_t dval = sl[v][xi * 64 + dw];
        unsigned char* dst = v ? xp2 : xp1;
        size_t off = ((((size_t)n * HWP) + (y + 1)) * HWP + (xi + 1)) * 256 + dw * 4;
        *(uint32_t*)(dst + off) = dval;
    }
}

__global__ void bz_pack_w_kernel(const float* __restrict__ w,
                                 unsigned char* __restrict__ wf8) {
    const int j   = blockIdx.x * 256 + threadIdx.x;
    const int pos = j >> 16;
    const int oc  = (j >> 8) & 255;
    const int ic  = j & 255;
    wf8[j] = sgn_fp8(w[((size_t)oc * ICH + ic) * 9 + pos]);
}

#if USE_TDM
__device__ __forceinline__ unsigned int lds_addr_of(const void* p) {
    return (unsigned int)(uintptr_t)(__attribute__((address_space(3))) const void*)p;
}

__device__ __forceinline__ void tdm_tile_load(
        unsigned int lds, unsigned long long ga,
        unsigned int tensor_d0, unsigned int tensor_d1,
        unsigned int tile_d0, unsigned int tile_d1, unsigned int tile_d2,
        unsigned long long d0_stride, unsigned long long d1_stride) {
    v4u g0; v8i g1; v4i g2; v4i g3; v8i g4;
    g0[0] = 1u;
    g0[1] = lds;
    g0[2] = (unsigned int)ga;
    g0[3] = (unsigned int)((ga >> 32) & 0x1FFFFFFu) | (2u << 30);
    g1[0] = (int)((1u << 20) | (4u << 22) | (3u << 25));
    g1[1] = (int)((tensor_d0 & 0xFFFFu) << 16);
    g1[2] = (int)(((tensor_d0 >> 16) & 0xFFFFu) | ((tensor_d1 & 0xFFFFu) << 16));
    g1[3] = (int)(((tensor_d1 >> 16) & 0xFFFFu) | (tile_d0 << 16));
    g1[4] = (int)((tile_d1 & 0xFFFFu) | (tile_d2 << 16));
    g1[5] = (int)(unsigned int)(d0_stride & 0xFFFFFFFFu);
    g1[6] = (int)((unsigned int)((d0_stride >> 32) & 0xFFFFu) |
                  (unsigned int)((d1_stride & 0xFFFFu) << 16));
    g1[7] = (int)(unsigned int)((d1_stride >> 16) & 0xFFFFFFFFu);
    g2[0] = 0x40000000; g2[1] = 0; g2[2] = 0; g2[3] = 0;
    g3[0] = 0; g3[1] = 0; g3[2] = 0; g3[3] = 0;
    g4[0] = 0; g4[1] = 0; g4[2] = 0; g4[3] = 0;
    g4[4] = 0; g4[5] = 0; g4[6] = 0; g4[7] = 0;
    __builtin_amdgcn_tensor_load_to_lds(g0, g1, g2, g3, g4, 0);
}
#endif

__global__ __launch_bounds__(256)
void bz_conv_wmma_kernel(const unsigned char* __restrict__ xp1,
                         const unsigned char* __restrict__ xp2,
                         const unsigned char* __restrict__ wf8,
                         const float* __restrict__ bias,
                         const float* __restrict__ w1p,
                         float* __restrict__ out) {
    __shared__ __align__(16) unsigned char sA[2][128 * ROWS];
    __shared__ __align__(16) unsigned char sB[2][2][112 * ROWS];

    const int t    = threadIdx.x;
    const int lane = t & 31;
    const int wv   = t >> 5;
    const int n    = blockIdx.x / 28;
    const int pbk  = blockIdx.x % 28;
    const int oh0  = pbk * 2;
    const int ocb  = blockIdx.y;

    const int m = lane & 15;
    const int h = lane >> 4;

    v8f c1[7] = {};
    v8f c2[7] = {};

#if USE_TDM
    if (wv == 0) {
        const int pos = 0, icb = 0, kh = 0, kw = 0;
        tdm_tile_load(lds_addr_of(&sA[0][0]),
                      (unsigned long long)(uintptr_t)(wf8 +
                          (((size_t)(pos * OCH + ocb * 128)) << 8) + icb * 128),
                      256, 256, 128, 128, 0, 256ull, 0ull);
        tdm_tile_load(lds_addr_of(&sB[0][0][0]),
                      (unsigned long long)(uintptr_t)(xp1 +
                          ((((size_t)n * HWP) + oh0 + kh) * HWP + kw) * 256 + icb * 128),
                      256, HWP, 128, HH, 2, 256ull, (unsigned long long)(HWP * 256));
        tdm_tile_load(lds_addr_of(&sB[0][1][0]),
                      (unsigned long long)(uintptr_t)(xp2 +
                          ((((size_t)n * HWP) + oh0 + kh) * HWP + kw) * 256 + icb * 128),
                      256, HWP, 128, HH, 2, 256ull, (unsigned long long)(HWP * 256));
    }
#endif

    for (int c = 0; c < 18; ++c) {
        const int cur = c & 1;
#if USE_TDM
        if (wv == 0) {
            const int cn = c + 1;
            if (cn < 18) {
                const int pos = cn >> 1, icb = cn & 1;
                const int kh = pos / 3, kw = pos % 3;
                const int nb = cn & 1;
                tdm_tile_load(lds_addr_of(&sA[nb][0]),
                              (unsigned long long)(uintptr_t)(wf8 +
                                  (((size_t)(pos * OCH + ocb * 128)) << 8) + icb * 128),
                              256, 256, 128, 128, 0, 256ull, 0ull);
                tdm_tile_load(lds_addr_of(&sB[nb][0][0]),
                              (unsigned long long)(uintptr_t)(xp1 +
                                  ((((size_t)n * HWP) + oh0 + kh) * HWP + kw) * 256 + icb * 128),
                              256, HWP, 128, HH, 2, 256ull, (unsigned long long)(HWP * 256));
                tdm_tile_load(lds_addr_of(&sB[nb][1][0]),
                              (unsigned long long)(uintptr_t)(xp2 +
                                  ((((size_t)n * HWP) + oh0 + kh) * HWP + kw) * 256 + icb * 128),
                              256, HWP, 128, HH, 2, 256ull, (unsigned long long)(HWP * 256));
                __builtin_amdgcn_s_wait_tensorcnt(3);
            } else {
                __builtin_amdgcn_s_wait_tensorcnt(0);
            }
        }
        __syncthreads();
#else
        {
            const int pos = c >> 1, icb = c & 1;
            const int kh = pos / 3, kw = pos % 3;
            {
                const int arow = t >> 1, ahalf = t & 1;
                const uint4* ga = (const uint4*)(wf8 +
                    (((size_t)(pos * OCH + ocb * 128 + arow)) << 8) + icb * 128 + ahalf * 64);
                uint4* la = (uint4*)(&sA[cur][arow * ROWS + ahalf * 64]);
                la[0] = ga[0]; la[1] = ga[1]; la[2] = ga[2]; la[3] = ga[3];
            }
            if (t < 224) {
                const int v = t / 112, r = t % 112;
                const int oh = oh0 + r / HH, ow = r % HH;
                const unsigned char* xs = v ? xp2 : xp1;
                const uint4* gb = (const uint4*)(xs +
                    ((((size_t)n * HWP) + oh + kh) * HWP + (ow + kw)) * 256 + icb * 128);
                uint4* lb = (uint4*)(&sB[cur][v][r * ROWS]);
#pragma unroll
                for (int q = 0; q < 8; ++q) lb[q] = gb[q];
            }
        }
        __syncthreads();
#endif

        v16i a;
        {
            const unsigned char* arow = &sA[cur][(wv * 16 + m) * ROWS];
#pragma unroll
            for (int d = 0; d < 8; ++d) {
                uint2 u = *(const uint2*)(arow + (d >> 1) * 32 + h * 16 + (d & 1) * 8);
                a[2 * d]     = (int)u.x;
                a[2 * d + 1] = (int)u.y;
            }
        }

#pragma unroll
        for (int j = 0; j < 7; ++j) {
            v16i b;
            const unsigned char* brow1 = &sB[cur][0][(j * 16 + m) * ROWS];
#pragma unroll
            for (int q = 0; q < 4; ++q) {
                uint4 u = *(const uint4*)(brow1 + q * 32 + h * 16);
                b[4 * q] = (int)u.x; b[4 * q + 1] = (int)u.y;
                b[4 * q + 2] = (int)u.z; b[4 * q + 3] = (int)u.w;
            }
            c1[j] = __builtin_amdgcn_wmma_f32_16x16x128_fp8_fp8(
                        a, b, (short)0, c1[j], false, false);

            const unsigned char* brow2 = &sB[cur][1][(j * 16 + m) * ROWS];
#pragma unroll
            for (int q = 0; q < 4; ++q) {
                uint4 u = *(const uint4*)(brow2 + q * 32 + h * 16);
                b[4 * q] = (int)u.x; b[4 * q + 1] = (int)u.y;
                b[4 * q + 2] = (int)u.z; b[4 * q + 3] = (int)u.w;
            }
            c2[j] = __builtin_amdgcn_wmma_f32_16x16x128_fp8_fp8(
                        a, b, (short)0, c2[j], false, false);
        }
        __syncthreads();
    }

    const float w1v = w1p[0];
    const int ocbase = ocb * 128 + wv * 16;
    float bvs[8];
#pragma unroll
    for (int r = 0; r < 8; ++r)
        bvs[r] = bias[ocbase + r + 8 * h] * (1.f + w1v);
#pragma unroll
    for (int j = 0; j < 7; ++j) {
        const int px = pbk * 112 + j * 16 + m;
#pragma unroll
        for (int r = 0; r < 8; ++r) {
            const int oc = ocbase + r + 8 * h;
            __builtin_nontemporal_store(c1[j][r] + w1v * c2[j][r] + bvs[r],
                                        &out[((size_t)n * OCH + oc) * HWPIX + px]);
        }
    }
}

extern "C" void kernel_launch(void* const* d_in, const int* in_sizes, int n_in,
                              void* d_out, int out_size, void* d_ws, size_t ws_size,
                              hipStream_t stream) {
    const float* x      = (const float*)d_in[0];
    const float* shift1 = (const float*)d_in[1];
    const float* shift2 = (const float*)d_in[2];
    const float* weight = (const float*)d_in[3];
    const float* bias   = (const float*)d_in[4];
    const float* w1     = (const float*)d_in[5];
    float* out = (float*)d_out;

    unsigned char* xp1 = (unsigned char*)d_ws;
    unsigned char* xp2 = xp1 + XPAD_BYTES;
    unsigned char* wf8 = xp2 + XPAD_BYTES;

    const unsigned int n16 = (2u * XPAD_BYTES) / 16u;
    bz_zero_kernel<<<(n16 + 255u) / 256u, 256, 0, stream>>>((uint4*)d_ws, n16);

    bz_pack_x_kernel<<<NB * HH, 256, 0, stream>>>(x, shift1, shift2, xp1, xp2);

    bz_pack_w_kernel<<<WF8_BYTES / 256, 256, 0, stream>>>(weight, wf8);

    dim3 grid(NB * 28, 2, 1);
    bz_conv_wmma_kernel<<<grid, 256, 0, stream>>>(xp1, xp2, wf8, bias, w1, out);
}
